// EdgeConvBlock_76338748719429
// MI455X (gfx1250) — matrix operand order repaired and checked by reading, not compiled, not run on hardware
//
#include <hip/hip_runtime.h>
#include <hip/hip_bf16.h>

typedef __bf16 v16bf __attribute__((ext_vector_type(16)));
typedef float  v8f   __attribute__((ext_vector_type(8)));

#define EPS 1e-5f
#define BB 128
#define NN 512
#define CC 64
#define KNBR 16
#define NK 8192      // NN * KNBR
#define SLICES 32    // sharded global stat accumulators

// ---------------- zero stats ----------------
__global__ void k_zero(float* p, int n) {
    int i = blockIdx.x * 256 + threadIdx.x;
    if (i < n) p[i] = 0.f;
}

// ---------------- KNN ----------------
// points: [B][2][512]  -> idx: [B][512][16]
__global__ __launch_bounds__(256) void k_knn(const float* __restrict__ pts,
                                             int* __restrict__ idx) {
    __shared__ float px[NN], py[NN];
    int b = blockIdx.x >> 1;
    int half = blockIdx.x & 1;
    const float* p = pts + (size_t)b * 2 * NN;
    for (int i = threadIdx.x; i < NN; i += 256) { px[i] = p[i]; py[i] = p[NN + i]; }
    __syncthreads();
    int n = half * 256 + threadIdx.x;
    float qx = px[n], qy = py[n];
    float dk[KNBR]; int ik[KNBR];
#pragma unroll
    for (int t = 0; t < KNBR; ++t) { dk[t] = 3.0e38f; ik[t] = 0; }
    for (int m = 0; m < NN; ++m) {
        if (m == n) continue;
        float dx = px[m] - qx, dy = py[m] - qy;
        float d = dx * dx + dy * dy;
        if (d < dk[KNBR - 1]) {
            float dd = d; int mi = m;
#pragma unroll
            for (int t = 0; t < KNBR; ++t) {
                if (dd < dk[t]) {
                    float td = dk[t]; int ti = ik[t];
                    dk[t] = dd; ik[t] = mi; dd = td; mi = ti;
                }
            }
        }
    }
    int* op = idx + ((size_t)b * NN + n) * KNBR;
#pragma unroll
    for (int t = 0; t < KNBR; ++t) op[t] = ik[t];
}

// WMMA A-fragment slot (16x32 bf16): row = lane%16
__device__ __forceinline__ int a_slot(int row, int kp /*0..31*/) {
    int g = kp >> 3;
    int e = ((g >> 1) << 3) + (kp & 7);
    int lane = ((g & 1) << 4) + row;
    return lane * 16 + e;
}
// WMMA B-fragment slot (32x16 bf16): col = lane%16, same K order as a_slot
__device__ __forceinline__ int b_slot(int col, int kp /*0..31*/) {
    int g = kp >> 3;
    int e = ((g >> 1) << 3) + (kp & 7);
    int lane = ((g & 1) << 4) + col;
    return lane * 16 + e;
}

// Butterfly-reduce one 16x16 D-tile's per-channel sum/sumsq across the 16 lanes
// that share each channel, then 2 lanes do 8 LDS atomics each.
__device__ __forceinline__ void tile_stats(v8f a, int lane, int cbase,
                                           float* osum, float* osq) {
    float s[8], q[8];
#pragma unroll
    for (int r = 0; r < 8; ++r) { s[r] = a[r]; q[r] = s[r] * s[r]; }
#pragma unroll
    for (int m = 8; m >= 1; m >>= 1) {
#pragma unroll
        for (int r = 0; r < 8; ++r) {
            s[r] += __shfl_xor(s[r], m, 16);
            q[r] += __shfl_xor(q[r], m, 16);
        }
    }
    if ((lane & 15) == 0) {
        int mb = (lane < 16) ? 0 : 8;
#pragma unroll
        for (int r = 0; r < 8; ++r) {
            atomicAdd(&osum[cbase + mb + r], s[r]);
            atomicAdd(&osq[cbase + mb + r], q[r]);
        }
    }
}

// ---------------- layer0: gather + concat + GEMM0 + stats ----------------
// y0[b][c][col] = sum_k w0[c][k] * X[k][col], col = n*16+kk, 128 cols per block
__global__ __launch_bounds__(256) void k_layer0(
    const float* __restrict__ feats, const int* __restrict__ idx,
    const float* __restrict__ w0, __bf16* __restrict__ y0,
    float* __restrict__ st0) {   // [SLICES][128] : 64 sum + 64 sumsq per slice
    __shared__ alignas(32) __bf16 ldsA[4 * 4 * 512];  // 16KB
    __shared__ alignas(32) __bf16 ldsB[4 * 8 * 512];  // 32KB
    __shared__ float ssum[64], ssq[64];
    int b = blockIdx.x >> 6;
    int chunk = blockIdx.x & 63;       // 128 columns = 8 points
    int slice = blockIdx.x & (SLICES - 1);
    int tid = threadIdx.x;
    if (tid < 64) { ssum[tid] = 0.f; ssq[tid] = 0.f; }

    // stage A: w0 (64 x 128)
    for (int i = tid; i < 64 * 128; i += 256) {
        int m = i >> 7, k = i & 127;
        int mt = m >> 4, kt = k >> 5;
        ldsA[(mt * 4 + kt) * 512 + a_slot(m & 15, k & 31)] = (__bf16)w0[i];
    }
    // stage B: gathered edge features, rows c 0..127, cols j 0..127
    const float* fB = feats + (size_t)b * CC * NN;
    const int* idxB = idx + (size_t)b * NN * KNBR + chunk * 128;
    int n0 = chunk * 8;
    for (int i = tid; i < 128 * 128; i += 256) {
        int c = i >> 7, j = i & 127;
        int nl = j >> 4, kk = j & 15;
        int n = n0 + nl;
        float v;
        if (c < 64) {
            v = fB[c * NN + n];
        } else {
            int m = idxB[nl * 16 + kk];
            int cc = c - 64;
            v = fB[cc * NN + m] - fB[cc * NN + n];
        }
        int kt = c >> 5;
        ldsB[(kt * 8 + nl) * 512 + b_slot(kk, c & 31)] = (__bf16)v;
    }
    __syncthreads();

    int wave = tid >> 5, lane = tid & 31;
    int nt = wave;  // 0..7
    v8f acc[4] = {};
#pragma unroll
    for (int kt = 0; kt < 4; ++kt) {
        v16bf bf = *(const v16bf*)&ldsB[(kt * 8 + nt) * 512 + lane * 16];
#pragma unroll
        for (int mt = 0; mt < 4; ++mt) {
            v16bf af = *(const v16bf*)&ldsA[(mt * 4 + kt) * 512 + lane * 16];
            acc[mt] = __builtin_amdgcn_wmma_f32_16x16x32_bf16(
                false, af, false, bf, (short)0, acc[mt], false, false);
        }
    }
    int col = chunk * 128 + nt * 16 + (lane & 15);
    int mb = (lane < 16) ? 0 : 8;
    size_t yb = (size_t)b * 64 * NK;
#pragma unroll
    for (int mt = 0; mt < 4; ++mt) {
#pragma unroll
        for (int r = 0; r < 8; ++r) {
            int c = mt * 16 + mb + r;
            y0[yb + (size_t)c * NK + col] = (__bf16)acc[mt][r];
        }
        tile_stats(acc[mt], lane, mt * 16, ssum, ssq);
    }
    __syncthreads();
    if (tid < 64) {
        atomicAdd(&st0[slice * 128 + tid], ssum[tid]);
        atomicAdd(&st0[slice * 128 + 64 + tid], ssq[tid]);
    }
}

// ---------------- layer1 / layer2: BN+ReLU on load, GEMM, stats ----------------
// Cin = KT*32 input channels, 128 output channels, 128 columns per block
template <int KT>
__global__ __launch_bounds__(256) void k_layerN(
    const __bf16* __restrict__ yin, const float* __restrict__ stin, // [SLICES][2*Cin]
    const float* __restrict__ gin, const float* __restrict__ bin,
    const float* __restrict__ w, __bf16* __restrict__ yout,
    float* __restrict__ stout) {                                    // [SLICES][256]
    constexpr int Cin = KT * 32;
    __shared__ alignas(32) __bf16 ldsA[8 * KT * 512];   // KT=4: 32KB
    __shared__ alignas(32) __bf16 ldsB[KT * 8 * 512];   // KT=4: 32KB
    __shared__ float sscale[Cin], sshift[Cin];
    __shared__ float osum[128], osq[128];
    int b = blockIdx.x >> 6;
    int chunk = blockIdx.x & 63;   // 128 columns
    int slice = blockIdx.x & (SLICES - 1);
    int tid = threadIdx.x;
    if (tid < 128) { osum[tid] = 0.f; osq[tid] = 0.f; }
    if (tid < Cin) {
        const float ic = 1.f / 1048576.f;  // B*N*K
        float s = 0.f, q = 0.f;
#pragma unroll 4
        for (int sl = 0; sl < SLICES; ++sl) {
            s += stin[sl * 2 * Cin + tid];
            q += stin[sl * 2 * Cin + Cin + tid];
        }
        float m = s * ic;
        float var = q * ic - m * m;
        float sc = gin[tid] * rsqrtf(var + EPS);
        sscale[tid] = sc;
        sshift[tid] = bin[tid] - m * sc;
    }
    __syncthreads();
    // stage A: w (128 x Cin)
    for (int i = tid; i < 128 * Cin; i += 256) {
        int m = i / Cin, k = i % Cin;
        int mt = m >> 4, kt = k >> 5;
        ldsA[(mt * KT + kt) * 512 + a_slot(m & 15, k & 31)] = (__bf16)w[i];
    }
    // stage B: relu(bn(yin)), rows c 0..Cin-1, cols j 0..127
    int cb = chunk * 128;
    const __bf16* yb = yin + (size_t)b * Cin * NK + cb;
    for (int i = tid; i < Cin * 128; i += 256) {
        int c = i >> 7, j = i & 127;
        float v = (float)yb[(size_t)c * NK + j] * sscale[c] + sshift[c];
        v = fmaxf(v, 0.f);
        int kt = c >> 5, nt = j >> 4;
        ldsB[(kt * 8 + nt) * 512 + b_slot(j & 15, c & 31)] = (__bf16)v;
    }
    __syncthreads();

    int wave = tid >> 5, lane = tid & 31;
    int nt = wave;  // 0..7 : wave owns one 16-col block, all 8 M-tiles
    v8f acc[8] = {};
#pragma unroll
    for (int kt = 0; kt < KT; ++kt) {
        v16bf bf = *(const v16bf*)&ldsB[(kt * 8 + nt) * 512 + lane * 16];
#pragma unroll
        for (int mt = 0; mt < 8; ++mt) {
            v16bf af = *(const v16bf*)&ldsA[(mt * KT + kt) * 512 + lane * 16];
            acc[mt] = __builtin_amdgcn_wmma_f32_16x16x32_bf16(
                false, af, false, bf, (short)0, acc[mt], false, false);
        }
    }
    int col = cb + nt * 16 + (lane & 15);
    int mb = (lane < 16) ? 0 : 8;
    size_t ob = (size_t)b * 128 * NK;
#pragma unroll
    for (int mt = 0; mt < 8; ++mt) {
#pragma unroll
        for (int r = 0; r < 8; ++r) {
            int c = mt * 16 + mb + r;
            yout[ob + (size_t)c * NK + col] = (__bf16)acc[mt][r];
        }
        tile_stats(acc[mt], lane, mt * 16, osum, osq);
    }
    __syncthreads();
    if (tid < 128) {
        atomicAdd(&stout[slice * 256 + tid], osum[tid]);
        atomicAdd(&stout[slice * 256 + 128 + tid], osq[tid]);
    }
}

// ---------------- shortcut GEMM + stats ----------------
__global__ __launch_bounds__(256) void k_short(const float* __restrict__ feats,
                                               const float* __restrict__ wsc,
                                               float* __restrict__ sc,
                                               float* __restrict__ st) { // [256]
    __shared__ float rs[2];
    int flat = blockIdx.x * 256 + threadIdx.x;
    int n = flat & 511;
    int o = (flat >> 9) & 127;
    int b = flat >> 16;
    if (threadIdx.x == 0) { rs[0] = 0.f; rs[1] = 0.f; }
    __syncthreads();
    const float* f = feats + (size_t)b * CC * NN;
    const float* wr = wsc + o * CC;
    float acc = 0.f;
#pragma unroll 8
    for (int c = 0; c < CC; ++c) acc += wr[c] * f[c * NN + n];
    sc[flat] = acc;
    atomicAdd(&rs[0], acc);
    atomicAdd(&rs[1], acc * acc);
    __syncthreads();
    if (threadIdx.x == 0) {
        atomicAdd(&st[o], rs[0]);
        atomicAdd(&st[128 + o], rs[1]);
    }
}

// ---------------- final: BN2+ReLU, mean over K, + BN'd shortcut, ReLU ----------
__global__ __launch_bounds__(256) void k_final(
    const __bf16* __restrict__ y2, const float* __restrict__ st2, // [SLICES][256]
    const float* __restrict__ g2, const float* __restrict__ b2,
    const float* __restrict__ sc, const float* __restrict__ stsc, // [256]
    const float* __restrict__ scg, const float* __restrict__ scb,
    float* __restrict__ out) {
    __shared__ float p[4];
    int flat = blockIdx.x * 256 + threadIdx.x;
    int o = (flat >> 9) & 127;
    if (threadIdx.x == 0) {
        const float ic = 1.f / 1048576.f;
        float s = 0.f, q = 0.f;
#pragma unroll 4
        for (int sl = 0; sl < SLICES; ++sl) {
            s += st2[sl * 256 + o];
            q += st2[sl * 256 + 128 + o];
        }
        float m = s * ic;
        float v = q * ic - m * m;
        float sf = g2[o] * rsqrtf(v + EPS);
        p[0] = sf; p[1] = b2[o] - m * sf;
        const float ic2 = 1.f / 65536.f;
        float m2 = stsc[o] * ic2;
        float v2 = stsc[128 + o] * ic2 - m2 * m2;
        float s2 = scg[o] * rsqrtf(v2 + EPS);
        p[2] = s2; p[3] = scb[o] - m2 * s2;
    }
    __syncthreads();
    const __bf16* yr = y2 + (size_t)flat * 16;
    float acc = 0.f;
#pragma unroll
    for (int k = 0; k < 16; ++k) {
        float v = (float)yr[k] * p[0] + p[1];
        acc += fmaxf(v, 0.f);
    }
    float fts = acc * (1.f / 16.f);
    float s = sc[flat] * p[2] + p[3];
    out[flat] = fmaxf(s + fts, 0.f);
}

extern "C" void kernel_launch(void* const* d_in, const int* in_sizes, int n_in,
                              void* d_out, int out_size, void* d_ws, size_t ws_size,
                              hipStream_t stream) {
    const float* points   = (const float*)d_in[0];
    const float* features = (const float*)d_in[1];
    const float* w0  = (const float*)d_in[2];
    const float* g0  = (const float*)d_in[3];
    const float* b0  = (const float*)d_in[4];
    const float* w1  = (const float*)d_in[5];
    const float* g1  = (const float*)d_in[6];
    const float* b1  = (const float*)d_in[7];
    const float* w2  = (const float*)d_in[8];
    const float* g2  = (const float*)d_in[9];
    const float* b2  = (const float*)d_in[10];
    const float* scw = (const float*)d_in[11];
    const float* scg = (const float*)d_in[12];
    const float* scb = (const float*)d_in[13];

    char* ws = (char*)d_ws;
    float* stats = (float*)ws;                    // 20736 floats total
    float* st0  = stats;                          // [32][128]
    float* st1  = stats + 4096;                   // [32][256]
    float* st2  = stats + 12288;                  // [32][256]
    float* stsc = stats + 20480;                  // [256]
    int*    idx = (int*)   (ws + 131072ull);      // 4 MB
    __bf16* y0  = (__bf16*)(ws + 4325376ull);     // [B][64][8192]  128 MB
    __bf16* y1  = (__bf16*)(ws + 138543104ull);   // [B][128][8192] 256 MB
    __bf16* y2  = (__bf16*)(ws + 406978560ull);   // [B][128][8192] 256 MB
    float*  sc  = (float*) (ws + 675414016ull);   // [B][128][512]  32 MB

    k_zero<<<81, 256, 0, stream>>>(stats, 20736);
    k_knn<<<BB * 2, 256, 0, stream>>>(points, idx);
    k_layer0<<<BB * 64, 256, 0, stream>>>(features, idx, w0, y0, st0);
    k_layerN<2><<<BB * 64, 256, 0, stream>>>(y0, st0, g0, b0, w1, y1, st1);
    k_layerN<4><<<BB * 64, 256, 0, stream>>>(y1, st1, g1, b1, w2, y2, st2);
    k_short<<<BB * 128 * 512 / 256, 256, 0, stream>>>(features, scw, sc, stsc);
    k_final<<<BB * 128 * 512 / 256, 256, 0, stream>>>(y2, st2, g2, b2, sc, stsc,
                                                      scg, scb, (float*)d_out);
}
